// RKHS_VNN_16277926052040
// MI455X (gfx1250) — matrix operand order repaired and checked by reading, not compiled, not run on hardware
//
#include <hip/hip_runtime.h>

#define USE_TDM 1

typedef __attribute__((ext_vector_type(16))) _Float16 v16h;
typedef __attribute__((ext_vector_type(8)))  _Float16 v8h;
typedef __attribute__((ext_vector_type(8)))  float    v8f;
typedef __attribute__((ext_vector_type(4)))  unsigned v4u;
typedef __attribute__((ext_vector_type(8)))  unsigned v8u;

// ---------------------------------------------------------------------------
// utility: 256-thread block reduction (deterministic)
// ---------------------------------------------------------------------------
__device__ __forceinline__ float block_reduce_256(float v, float* sm) {
  const int t = threadIdx.x;
  __syncthreads();
  sm[t] = v;
  __syncthreads();
#pragma unroll
  for (int s = 128; s > 0; s >>= 1) {
    if (t < s) sm[t] += sm[t + s];
    __syncthreads();
  }
  return sm[0];
}

// ---------------------------------------------------------------------------
// Tensor Data Mover: load a (32 x 64) f16 tile of wt into LDS with the D#
// pad feature producing a 72-half row stride (bank-conflict-free b128 reads).
// Issued once per wave (scalar instruction); tracked by TENSORcnt.
// ---------------------------------------------------------------------------
__device__ __forceinline__ void tdm_issue_b(const _Float16* gsrc,
                                            _Float16* ldst, int Kp,
                                            int COUTp) {
  unsigned long long ga = (unsigned long long)(const void*)gsrc;
  unsigned ldsb = (unsigned)(unsigned long long)(void*)ldst;
  v4u g0;
  g0.x = 1u;                                                // count=1
  g0.y = ldsb;                                              // lds_addr
  g0.z = (unsigned)(ga & 0xffffffffu);                      // global_addr lo
  g0.w = (unsigned)((ga >> 32) & 0x01ffffffu) | (2u << 30); // addr hi | type=2
  v8u g1;
  // data_size=2B | pad_enable | pad_interval=32DW(code 4) | pad_amount=4DW(3)
  g1.s0 = (1u << 16) | (1u << 20) | (4u << 22) | (3u << 25);
  g1.s1 = ((unsigned)Kp & 0xffffu) << 16;                   // tensor_dim0 lo
  g1.s2 = ((unsigned)Kp >> 16) | (((unsigned)COUTp & 0xffffu) << 16);
  g1.s3 = ((unsigned)COUTp >> 16) | (64u << 16);            // tile_dim0=64
  g1.s4 = 32u;                                              // tile_dim1=32
  g1.s5 = (unsigned)Kp;                                     // dim0 stride
  g1.s6 = 0u;
  g1.s7 = 0u;
  asm volatile("tensor_load_to_lds %0, %1" ::"s"(g0), "s"(g1) : "memory");
}

// ---------------------------------------------------------------------------
// x (B,C,D,H,W) f32  ->  xlast (B,D,H,W,C) f32
// ---------------------------------------------------------------------------
__global__ __launch_bounds__(256) void k_to_channels_last(
    const float* __restrict__ x, float* __restrict__ y,
    int B, int C, long S /* = D*H*W */) {
  long n = (long)B * C * S;
  long i = (long)blockIdx.x * 256 + threadIdx.x;
  if (i >= n) return;
  long s = i % S;
  long t = i / S;
  int  c = (int)(t % C);
  int  b = (int)(t / C);
  y[(b * S + s) * C + c] = x[i];
}

// ---------------------------------------------------------------------------
// W_eff build:  wt[o, kp] = sum_p proj[p, c*27 + n] * alpha[p, o]
//   kp = n*C + c  (im2col order: neighbor-major, channel-minor)
//   wt stored transposed (COUTp rows of Kp), Kp mult-of-64, COUTp mult-of-32,
//   padding rows/cols zero-filled so the conv kernel needs no load guards.
// ---------------------------------------------------------------------------
__global__ __launch_bounds__(256) void k_build_wt(
    const float* __restrict__ proj,   // (512, K)
    const float* __restrict__ alpha,  // (512, COUT)
    _Float16* __restrict__ wt,        // (COUTp, Kp)
    int C, int K, int Kp, int COUT) {
  int kp = blockIdx.x * 256 + threadIdx.x;
  int o  = blockIdx.y;                // 0..COUTp-1
  if (kp >= Kp) return;
  float s = 0.0f;
  if (kp < K && o < COUT) {
    int c = kp % C;
    int n = kp / C;
    int feat = c * 27 + n;
    for (int p = 0; p < 512; ++p)
      s += proj[(size_t)p * K + feat] * alpha[(size_t)p * COUT + o];
  }
  wt[(size_t)o * Kp + kp] = (_Float16)s;
}

// ---------------------------------------------------------------------------
// Fused im2col + GEMM + clamp(1+x,0), FP16 WMMA, f32 accum.
//   in : (B,D,H,W,C) f32 channel-last
//   wt : (COUTp, Kp) f16  (W_eff transposed, zero-padded)
//   out: (M, COUT) f32,  M = B*D*H*W   (pre-pool, post-clamp)
// Block: 256 thr = 8 waves (4 M-groups x 2 N-groups).
// Block tile: 128 voxels x 32 channels; wave tile 32x16 (2 accumulators).
// K step 64 -> 4 v_wmma_f32_16x16x32_f16 per wave per staging round.
// B tiles double-buffered through the Tensor Data Mover: the DMA of tile
// k0+64 overlaps the im2col gather + WMMA of tile k0.
// ---------------------------------------------------------------------------
template <int C>
__global__ __launch_bounds__(256) void k_conv_wmma(
    const float* __restrict__ in, const _Float16* __restrict__ wt,
    float* __restrict__ out,
    int B, int D, int H, int W, int COUT, int COUTp, int K, int Kp) {
  const int M  = B * D * H * W;
  const int m0 = blockIdx.x * 128;
  const int o0 = blockIdx.y * 32;     // N base (block)

  constexpr int LDA = 72;  // f16/row: 144B (16B-aligned, conflict-free b128)
  constexpr int LDB = 72;
  __shared__ __attribute__((aligned(16))) _Float16 sA[128 * LDA];
  __shared__ __attribute__((aligned(16))) _Float16 sB[2][32 * LDB];
  __shared__ int sCoord[128];

  const int t    = threadIdx.x;
  const int lane = t & 31;
  const int wave = t >> 5;
  const int wgM  = wave & 3;          // 0..3 : M group (32 rows)
  const int wgN  = wave >> 2;         // 0..1 : N group (16 cols)

  // ---- one-time voxel decomposition (packed b|d|h|w) --------------------
  if (t < 128) {
    int v  = m0 + t;
    int pk = -1;
    if (v < M) {
      int w  = v % W;  int t1 = v / W;
      int h  = t1 % H; int t2 = t1 / H;
      int d  = t2 % D; int b  = t2 / D;
      pk = (b << 30) | (d << 20) | (h << 10) | w;
    }
    sCoord[t] = pk;
  }

#if USE_TDM
  // prime the pipeline: DMA of the first B tile overlaps sCoord + first gather
  if (wave == 0)
    tdm_issue_b(wt + (size_t)o0 * Kp, &sB[0][0], Kp, COUTp);
#endif
  __syncthreads();

  v8f acc0 = {};
  v8f acc1 = {};

  const int kk    = t & 63;   // column this thread stages (K-invariant)
  const int rbase = t >> 6;   // 0..3

  int par = 0;
  for (int k0 = 0; k0 < Kp; k0 += 64, par ^= 1) {
    const bool has_next = (k0 + 64) < Kp;
#if USE_TDM
    // ---- issue DMA of the NEXT B tile into the alternate buffer --------
    if (wave == 0 && has_next)
      tdm_issue_b(wt + (size_t)o0 * Kp + (k0 + 64), &sB[par ^ 1][0], Kp,
                  COUTp);
#else
    // ---- B tile fallback: per-thread loads (wt fully padded) ----------
#pragma unroll
    for (int i = 0; i < 8; ++i) {
      int o = (t >> 6) + i * 4;      // 0..31
      sB[par][o * LDB + kk] = wt[(size_t)(o0 + o) * Kp + k0 + kk];
    }
#endif

    // ---- A tile: 128 x 64 im2col gather (k-decode hoisted) ------------
    {
      int  k    = k0 + kk;
      bool kval = (k < K);
      int c = 0, dz = 0, dy = 0, dx = 0;
      if (kval) {
        c = k % C;
        int n  = k / C;
        dz = n / 9;
        int n2 = n % 9;
        dy = n2 / 3;
        dx = n2 % 3;
      }
#pragma unroll
      for (int i = 0; i < 32; ++i) {
        int r  = rbase + i * 4;
        int pk = sCoord[r];
        float val = 0.0f;
        if (kval && pk >= 0) {
          int w = pk & 1023, h = (pk >> 10) & 1023;
          int d = (pk >> 20) & 1023, b = (pk >> 30) & 1;
          int dd = d + dz - 1, hh = h + dy - 1, ww = w + dx - 1;
          if ((unsigned)dd < (unsigned)D && (unsigned)hh < (unsigned)H &&
              (unsigned)ww < (unsigned)W)
            val = in[((size_t)((b * D + dd) * H + hh) * W + ww) * C + c];
        }
        sA[r * LDA + kk] = (_Float16)val;
      }
    }

#if USE_TDM
    // current tile must have landed: with one newer DMA in flight wait <=1,
    // TENSORcnt completes in order so the older transfer is then done.
    if (wave == 0) {
      if (has_next) __builtin_amdgcn_s_wait_tensorcnt(1);
      else          __builtin_amdgcn_s_wait_tensorcnt(0);
    }
#endif
    __syncthreads();

    // ---- 4 WMMAs per wave ---------------------------------------------
    {
      const int m     = lane & 15;
      const int akoff = (lane >> 4) * 8;    // A: K split {0..7,16..23}
      const int bkoff = (lane >> 4) * 8;    // B: K split {0..7,16..23}
      const _Float16* Br  = &sB[par][(wgN * 16 + m) * LDB];
      const _Float16* Ar0 = &sA[(wgM * 32 + m) * LDA];
      const _Float16* Ar1 = &sA[(wgM * 32 + 16 + m) * LDA];

      v8h b0l = *(const v8h*)(Br + bkoff);
      v8h b0h = *(const v8h*)(Br + bkoff + 16);
      v8h b1l = *(const v8h*)(Br + 32 + bkoff);
      v8h b1h = *(const v8h*)(Br + 32 + bkoff + 16);
      v16h bv0 = __builtin_shufflevector(b0l, b0h, 0, 1, 2, 3, 4, 5, 6, 7, 8,
                                         9, 10, 11, 12, 13, 14, 15);
      v16h bv1 = __builtin_shufflevector(b1l, b1h, 0, 1, 2, 3, 4, 5, 6, 7, 8,
                                         9, 10, 11, 12, 13, 14, 15);

      v8h a;
      v8h ah;
      v16h av;
      a  = *(const v8h*)(Ar0 + akoff);
      ah = *(const v8h*)(Ar0 + akoff + 16);
      av = __builtin_shufflevector(a, ah, 0, 1, 2, 3, 4, 5, 6, 7, 8, 9, 10,
                                   11, 12, 13, 14, 15);
      acc0 = __builtin_amdgcn_wmma_f32_16x16x32_f16(false, av, false, bv0,
                                                    (short)0, acc0, false,
                                                    false);
      a  = *(const v8h*)(Ar0 + 32 + akoff);
      ah = *(const v8h*)(Ar0 + 32 + akoff + 16);
      av = __builtin_shufflevector(a, ah, 0, 1, 2, 3, 4, 5, 6, 7, 8, 9, 10,
                                   11, 12, 13, 14, 15);
      acc0 = __builtin_amdgcn_wmma_f32_16x16x32_f16(false, av, false, bv1,
                                                    (short)0, acc0, false,
                                                    false);
      a  = *(const v8h*)(Ar1 + akoff);
      ah = *(const v8h*)(Ar1 + akoff + 16);
      av = __builtin_shufflevector(a, ah, 0, 1, 2, 3, 4, 5, 6, 7, 8, 9, 10,
                                   11, 12, 13, 14, 15);
      acc1 = __builtin_amdgcn_wmma_f32_16x16x32_f16(false, av, false, bv0,
                                                    (short)0, acc1, false,
                                                    false);
      a  = *(const v8h*)(Ar1 + 32 + akoff);
      ah = *(const v8h*)(Ar1 + 32 + akoff + 16);
      av = __builtin_shufflevector(a, ah, 0, 1, 2, 3, 4, 5, 6, 7, 8, 9, 10,
                                   11, 12, 13, 14, 15);
      acc1 = __builtin_amdgcn_wmma_f32_16x16x32_f16(false, av, false, bv1,
                                                    (short)0, acc1, false,
                                                    false);
    }
    __syncthreads();
  }

  // ---- epilogue: clamp(1+x, 0), scatter --------------------------------
  const int n  = lane & 15;
  const int mh = (lane >> 4) * 8;
  const int og = o0 + wgN * 16 + n;
  if (og < COUT) {
#pragma unroll
    for (int j = 0; j < 8; ++j) {
      int v0 = m0 + wgM * 32 + mh + j;
      int v1 = v0 + 16;
      if (v0 < M) out[(size_t)v0 * COUT + og] = fmaxf(1.0f + acc0[j], 0.0f);
      if (v1 < M) out[(size_t)v1 * COUT + og] = fmaxf(1.0f + acc1[j], 0.0f);
    }
  }
}

// ---------------------------------------------------------------------------
// MaxPool (1,2,2), channel-last: (B,D,H,W,O) -> (B,D,H/2,W/2,O)
// ---------------------------------------------------------------------------
__global__ __launch_bounds__(256) void k_pool_hw(
    const float* __restrict__ in, float* __restrict__ out,
    int B, int D, int H, int W, int O) {
  int Ho = H >> 1, Wo = W >> 1;
  long n = (long)B * D * Ho * Wo * O;
  long i = (long)blockIdx.x * 256 + threadIdx.x;
  if (i >= n) return;
  int  o  = (int)(i % O);   long s = i / O;
  int  wo = (int)(s % Wo);  s /= Wo;
  int  ho = (int)(s % Ho);  s /= Ho;
  int  d  = (int)(s % D);
  int  b  = (int)(s / D);
  size_t base = ((size_t)((b * D + d) * H + 2 * ho) * W + 2 * wo) * O + o;
  float v0 = in[base];
  float v1 = in[base + O];
  float v2 = in[base + (size_t)W * O];
  float v3 = in[base + (size_t)W * O + O];
  out[i] = fmaxf(fmaxf(v0, v1), fmaxf(v2, v3));
}

// ---------------------------------------------------------------------------
// BatchNorm stats (train-mode, population var), deterministic.
// ---------------------------------------------------------------------------
__global__ __launch_bounds__(256) void k_bn_stats(
    const float* __restrict__ x, const float* __restrict__ gamma,
    const float* __restrict__ beta, float* __restrict__ ss,
    long Nvox, int O) {
  __shared__ float sm[256];
  const int o = blockIdx.x;
  float s = 0.0f, s2 = 0.0f;
  for (long i = threadIdx.x; i < Nvox; i += 256) {
    float v = x[i * O + o];
    s += v;
    s2 += v * v;
  }
  float ts  = block_reduce_256(s, sm);
  float ts2 = block_reduce_256(s2, sm);
  if (threadIdx.x == 0) {
    float mean = ts / (float)Nvox;
    float var  = ts2 / (float)Nvox - mean * mean;
    float sc   = gamma[o] * rsqrtf(var + 1e-5f);
    ss[o]     = sc;
    ss[O + o] = beta[o] - mean * sc;
  }
}

__global__ __launch_bounds__(256) void k_bn_apply(
    float* __restrict__ x, const float* __restrict__ ss, long N, int O) {
  long i = (long)blockIdx.x * 256 + threadIdx.x;
  if (i >= N) return;
  int o = (int)(i % O);
  x[i] = x[i] * ss[o] + ss[O + o];
}

// ---------------------------------------------------------------------------
// FC head
// ---------------------------------------------------------------------------
__global__ __launch_bounds__(256) void k_fc1(
    const float* __restrict__ act,  // (2, 144, 384) channel-last
    const float* __restrict__ w1,   // (512, 55296) channel-first features
    const float* __restrict__ b1, float* __restrict__ out /* (2,512) */) {
  __shared__ float sm[256];
  const int j = blockIdx.x & 511;
  const int b = blockIdx.x >> 9;
  const float* wr = w1 + (size_t)j * 55296;
  float s = 0.0f;
  for (int f = threadIdx.x; f < 55296; f += 256) {
    int c  = f / 144;
    int sp = f % 144;
    s += wr[f] * act[((size_t)b * 144 + sp) * 384 + c];
  }
  s = block_reduce_256(s, sm);
  if (threadIdx.x == 0) out[b * 512 + j] = fmaxf(s + b1[j], 0.0f);
}

__global__ __launch_bounds__(256) void k_fc_dense(
    const float* __restrict__ in, const float* __restrict__ w,
    const float* __restrict__ bias, float* __restrict__ out,
    int IN, int OUT, int relu) {
  __shared__ float sm[256];
  const int j = blockIdx.x % OUT;
  const int b = blockIdx.x / OUT;
  const float* wr = w + (size_t)j * IN;
  const float* xr = in + (size_t)b * IN;
  float s = 0.0f;
  for (int f = threadIdx.x; f < IN; f += 256) s += wr[f] * xr[f];
  s = block_reduce_256(s, sm);
  if (threadIdx.x == 0) {
    s += bias[j];
    out[b * OUT + j] = relu ? fmaxf(s, 0.0f) : s;
  }
}

// ---------------------------------------------------------------------------
// host launcher
// ---------------------------------------------------------------------------
static void launch_conv(int C, dim3 g, hipStream_t st, const float* in,
                        const _Float16* wt, float* out, int B, int D, int H,
                        int W, int CO, int COp, int K, int Kp) {
  switch (C) {
    case 3:   k_conv_wmma<3><<<g, 256, 0, st>>>(in, wt, out, B, D, H, W, CO, COp, K, Kp); break;
    case 24:  k_conv_wmma<24><<<g, 256, 0, st>>>(in, wt, out, B, D, H, W, CO, COp, K, Kp); break;
    case 48:  k_conv_wmma<48><<<g, 256, 0, st>>>(in, wt, out, B, D, H, W, CO, COp, K, Kp); break;
    case 96:  k_conv_wmma<96><<<g, 256, 0, st>>>(in, wt, out, B, D, H, W, CO, COp, K, Kp); break;
    case 192: k_conv_wmma<192><<<g, 256, 0, st>>>(in, wt, out, B, D, H, W, CO, COp, K, Kp); break;
  }
}

extern "C" void kernel_launch(void* const* d_in, const int* in_sizes, int n_in,
                              void* d_out, int out_size, void* d_ws,
                              size_t ws_size, hipStream_t stream) {
  (void)in_sizes; (void)n_in; (void)out_size; (void)ws_size;
  static const int CIN_[5]  = {3, 24, 48, 96, 192};
  static const int COUT_[5] = {24, 48, 96, 192, 384};

  const float* x = (const float*)d_in[0];
  const float *p[5], *a[5], *g[5], *q[5];
  for (int i = 0; i < 5; ++i) {
    p[i] = (const float*)d_in[1 + 4 * i];
    a[i] = (const float*)d_in[2 + 4 * i];
    g[i] = (const float*)d_in[3 + 4 * i];
    q[i] = (const float*)d_in[4 + 4 * i];
  }
  const float* w1 = (const float*)d_in[21];
  const float* b1 = (const float*)d_in[22];
  const float* w2 = (const float*)d_in[23];
  const float* b2 = (const float*)d_in[24];
  const float* w3 = (const float*)d_in[25];
  const float* b3 = (const float*)d_in[26];
  float* logits = (float*)d_out;

  // ---- workspace carve-up (256B aligned) ------------------------------
  char*  ws  = (char*)d_ws;
  size_t off = 0;
  auto carve = [&](size_t bytes) {
    size_t r = off;
    off += (bytes + 255) & ~(size_t)255;
    return r;
  };
  const long SP0 = 16L * 112 * 112;
  float*    xlast = (float*)(ws + carve(2L * SP0 * 3 * 4));            // 4.8 MB
  float*    bufA  = (float*)(ws + carve(2L * SP0 * 24 * 4));           // 38.5 MB
  float*    bufB  = (float*)(ws + carve(2L * 16 * 56 * 56 * 24 * 4));  // 9.6 MB
  _Float16* wt    = (_Float16*)(ws + carve(384L * 5184 * 2));          // 4.0 MB
  float*    ss    = (float*)(ws + carve(384L * 2 * 4));
  float*    h1    = (float*)(ws + carve(2L * 512 * 4));
  float*    h2    = (float*)(ws + carve(2L * 256 * 4));

  // ---- input -> channel-last ------------------------------------------
  {
    long n = 2L * 3 * SP0;
    k_to_channels_last<<<dim3((unsigned)((n + 255) / 256)), 256, 0, stream>>>(
        x, xlast, 2, 3, SP0);
  }

  // ---- 5 poly layers ---------------------------------------------------
  int D = 16, H = 112, W = 112;
  const float* cur = xlast;
  for (int i = 0; i < 5; ++i) {
    const int C   = CIN_[i];
    const int CO  = COUT_[i];
    const int K   = C * 27;
    const int Kp  = (K + 63) & ~63;    // K padded to multiple of 64
    const int COp = (CO + 31) & ~31;   // COUT padded to multiple of 32

    // W_eff = proj^T @ alpha (permuted, transposed, zero-padded, f16)
    k_build_wt<<<dim3((Kp + 255) / 256, COp), 256, 0, stream>>>(p[i], a[i], wt,
                                                                C, K, Kp, CO);

    // fused im2col + WMMA GEMM + clamp(1+x,0)
    const int M = 2 * D * H * W;
    dim3 grid((M + 127) / 128, COp / 32);
    launch_conv(C, grid, stream, cur, wt, bufA, 2, D, H, W, CO, COp, K, Kp);

    // maxpool (1,2,2)
    const int Ho = H / 2, Wo = W / 2;
    long nPool = 2L * D * Ho * Wo * CO;
    k_pool_hw<<<dim3((unsigned)((nPool + 255) / 256)), 256, 0, stream>>>(
        bufA, bufB, 2, D, H, W, CO);

    // batchnorm (train-mode batch statistics)
    long nVox = 2L * D * Ho * Wo;
    k_bn_stats<<<dim3(CO), 256, 0, stream>>>(bufB, g[i], q[i], ss, nVox, CO);
    k_bn_apply<<<dim3((unsigned)((nPool + 255) / 256)), 256, 0, stream>>>(
        bufB, ss, nPool, CO);

    H = Ho;
    W = Wo;
    cur = bufB;
  }

  // ---- classifier head -------------------------------------------------
  k_fc1<<<dim3(2 * 512), 256, 0, stream>>>(bufB, w1, b1, h1);
  k_fc_dense<<<dim3(2 * 256), 256, 0, stream>>>(h1, w2, b2, h2, 512, 256, 1);
  k_fc_dense<<<dim3(2 * 101), 256, 0, stream>>>(h2, w3, b3, logits, 256, 101, 0);
}
